// GQAAttention_89618787599005
// MI455X (gfx1250) — matrix operand order repaired and checked by reading, not compiled, not run on hardware
//
#include <hip/hip_runtime.h>
#include <stdint.h>

constexpr int kDim   = 1024;
constexpr int kHeads = 16;
constexpr int kKV    = 4;
constexpr int kHd    = 64;
constexpr int kSeq   = 2048;
constexpr int kBatch = 2;
constexpr int kM     = kBatch * kSeq;     // 4096 tokens
constexpr float kScale = 0.125f;          // 64^-0.5

typedef __attribute__((ext_vector_type(16))) __bf16 v16bf;
typedef __attribute__((ext_vector_type(8)))  float  v8f;

union Frag {
    v16bf v;
    uint16_t u[16];
    uint4 q[2];
};

__device__ __forceinline__ uint16_t f2bf(float f) {
    uint32_t x = __float_as_uint(f);
    x += 0x7FFFu + ((x >> 16) & 1u);      // round-to-nearest-even
    return (uint16_t)(x >> 16);
}

__device__ __forceinline__ v8f wmma_bf16(v16bf a, v16bf b, v8f c) {
    return __builtin_amdgcn_wmma_f32_16x16x32_bf16(false, a, false, b, (short)0, c,
                                                   false, false);
}

// A fragment 16(M)x32(K): lane(l&15)=row; K(e) = e + (e>=8?8:0) + 8*(l>=16)
__device__ __forceinline__ v16bf load_a(const uint16_t* __restrict__ base, int ld) {
    const int lane = threadIdx.x & 31;
    const uint16_t* p = base + (lane & 15) * ld + (lane >> 4) * 8;
    Frag f;
    f.q[0] = *(const uint4*)(p);
    f.q[1] = *(const uint4*)(p + 16);
    return f.v;
}

// B fragment 32(K)x16(N), column n contiguous along K at base + n*ld.
// K(e) = e + (e>=8?8:0) + 8*(l>=16)
__device__ __forceinline__ v16bf load_b_colK(const uint16_t* __restrict__ base, int ld) {
    const int lane = threadIdx.x & 31;
    const uint16_t* p = base + (lane & 15) * ld + (lane >> 4) * 8;
    Frag f;
    f.q[0] = *(const uint4*)(p);
    f.q[1] = *(const uint4*)(p + 16);
    return f.v;
}

__global__ void __launch_bounds__(256) k_cvt4(const float4* __restrict__ in,
                                              uint16_t* __restrict__ out, int n4) {
    int i = blockIdx.x * 256 + threadIdx.x;
    if (i < n4) {
        float4 v = in[i];
        union { uint16_t u[4]; uint2 w; } o;
        o.u[0] = f2bf(v.x); o.u[1] = f2bf(v.y);
        o.u[2] = f2bf(v.z); o.u[3] = f2bf(v.w);
        *(uint2*)(out + (size_t)i * 4) = o.w;
    }
}

// ---------------- GEMM core: wave computes 32(M) x 64(N), double-buffered ----
#define GEMM_BODY(APTR, WPTR)                                                   \
    v8f acc[2][4];                                                              \
    _Pragma("unroll") for (int s = 0; s < 2; ++s)                               \
        _Pragma("unroll") for (int t = 0; t < 4; ++t)                           \
            _Pragma("unroll") for (int i = 0; i < 8; ++i) acc[s][t][i] = 0.f;   \
    const uint16_t* ap = (APTR);                                                \
    v16bf a0 = load_a(ap, kDim);                                                \
    v16bf a1 = load_a(ap + (size_t)16 * kDim, kDim);                            \
    v16bf b0 = load_b_colK((WPTR) + (size_t)(nb +  0) * kDim, kDim);            \
    v16bf b1 = load_b_colK((WPTR) + (size_t)(nb + 16) * kDim, kDim);            \
    v16bf b2 = load_b_colK((WPTR) + (size_t)(nb + 32) * kDim, kDim);            \
    v16bf b3 = load_b_colK((WPTR) + (size_t)(nb + 48) * kDim, kDim);            \
    for (int k = 0; k < kDim; k += 32) {                                        \
        const int kn = (k + 32) & (kDim - 1);                                   \
        v16bf na0 = load_a(ap + kn, kDim);                                      \
        v16bf na1 = load_a(ap + (size_t)16 * kDim + kn, kDim);                  \
        v16bf nb0 = load_b_colK((WPTR) + (size_t)(nb +  0) * kDim + kn, kDim);  \
        v16bf nb1 = load_b_colK((WPTR) + (size_t)(nb + 16) * kDim + kn, kDim);  \
        v16bf nb2 = load_b_colK((WPTR) + (size_t)(nb + 32) * kDim + kn, kDim);  \
        v16bf nb3 = load_b_colK((WPTR) + (size_t)(nb + 48) * kDim + kn, kDim);  \
        acc[0][0] = wmma_bf16(a0, b0, acc[0][0]);                               \
        acc[1][0] = wmma_bf16(a1, b0, acc[1][0]);                               \
        acc[0][1] = wmma_bf16(a0, b1, acc[0][1]);                               \
        acc[1][1] = wmma_bf16(a1, b1, acc[1][1]);                               \
        acc[0][2] = wmma_bf16(a0, b2, acc[0][2]);                               \
        acc[1][2] = wmma_bf16(a1, b2, acc[1][2]);                               \
        acc[0][3] = wmma_bf16(a0, b3, acc[0][3]);                               \
        acc[1][3] = wmma_bf16(a1, b3, acc[1][3]);                               \
        a0 = na0; a1 = na1; b0 = nb0; b1 = nb1; b2 = nb2; b3 = nb3;             \
    }

// Q = X @ Wq^T, scattered into [B, H, N, D] bf16
__global__ void __launch_bounds__(256) k_gemm_q(const uint16_t* __restrict__ X,
                                                const uint16_t* __restrict__ W,
                                                uint16_t* __restrict__ Q) {
    const int warp = threadIdx.x >> 5, lane = threadIdx.x & 31;
    const int m0 = (blockIdx.x * 8 + warp) * 32;
    const int nb = blockIdx.y * 64;
    GEMM_BODY(X + (size_t)m0 * kDim, W)
    const int hi = lane >> 4, nl = lane & 15;
#pragma unroll
    for (int s = 0; s < 2; ++s)
#pragma unroll
        for (int t = 0; t < 4; ++t) {
            const int col = nb + t * 16 + nl;
            const int h = col >> 6, d = col & 63;
#pragma unroll
            for (int r = 0; r < 8; ++r) {
                const int gm = m0 + s * 16 + r + 8 * hi;
                const int bb = gm >> 11, tok = gm & (kSeq - 1);
                Q[(size_t)(((bb * kHeads) + h) * kSeq + tok) * kHd + d] =
                    f2bf(acc[s][t][r]);
            }
        }
}

// KV = X @ Wkv^T -> K [B, KV, N, D] bf16 and V transposed Vt [B, KV, D, N] bf16
__global__ void __launch_bounds__(256) k_gemm_kv(const uint16_t* __restrict__ X,
                                                 const uint16_t* __restrict__ W,
                                                 uint16_t* __restrict__ Kd,
                                                 uint16_t* __restrict__ Vt) {
    const int warp = threadIdx.x >> 5, lane = threadIdx.x & 31;
    const int m0 = (blockIdx.x * 8 + warp) * 32;
    const int nb = blockIdx.y * 64;
    GEMM_BODY(X + (size_t)m0 * kDim, W)
    const int hi = lane >> 4, nl = lane & 15;
#pragma unroll
    for (int s = 0; s < 2; ++s)
#pragma unroll
        for (int t = 0; t < 4; ++t) {
            const int col = nb + t * 16 + nl;   // 0..511 = pair*256 + kv*64 + d
            const int pair = col >> 8;
            const int kvh = (col >> 6) & 3;
            const int d = col & 63;
#pragma unroll
            for (int r = 0; r < 8; ++r) {
                const int gm = m0 + s * 16 + r + 8 * hi;
                const int bb = gm >> 11, tok = gm & (kSeq - 1);
                const uint16_t val = f2bf(acc[s][t][r]);
                if (pair == 0)
                    Kd[(size_t)(((bb * kKV) + kvh) * kSeq + tok) * kHd + d] = val;
                else
                    Vt[(size_t)(((bb * kKV) + kvh) * kHd + d) * kSeq + tok] = val;
            }
        }
}

// Flash attention: one wave per (b, h, 16-query tile). Output [B, N, H*D] bf16.
__global__ void __launch_bounds__(256) k_attn(const uint16_t* __restrict__ Q,
                                              const uint16_t* __restrict__ K,
                                              const uint16_t* __restrict__ Vt,
                                              uint16_t* __restrict__ O) {
    __shared__ __align__(64) uint16_t pbuf[8][16][32];
    const int warp = threadIdx.x >> 5, lane = threadIdx.x & 31;
    const int wid = blockIdx.x * 8 + warp;
    const int qt = wid & 127;            // N/16 tiles
    const int h = (wid >> 7) & 15;
    const int b = wid >> 11;
    const int kvh = h >> 2;              // GQA: head h -> kv head h/4
    const uint16_t* qp = Q + (size_t)(((b * kHeads) + h) * kSeq + qt * 16) * kHd;
    const uint16_t* kp = K + (size_t)((b * kKV) + kvh) * kSeq * kHd;
    const uint16_t* vp = Vt + (size_t)((b * kKV) + kvh) * kHd * kSeq;

    const v16bf aq0 = load_a(qp, kHd);          // D = 0..31
    const v16bf aq1 = load_a(qp + 32, kHd);     // D = 32..63

    float mrow[8], lrow[8];
    v8f oacc[4];
#pragma unroll
    for (int r = 0; r < 8; ++r) { mrow[r] = -3.0e38f; lrow[r] = 0.f; }
#pragma unroll
    for (int t = 0; t < 4; ++t)
#pragma unroll
        for (int i = 0; i < 8; ++i) oacc[t][i] = 0.f;

    const int hi = lane >> 4, nl = lane & 15;

    // prime K fragments for chunk 0
    v16bf bk0 = load_b_colK(kp + 0 * kHd, kHd);
    v16bf bk1 = load_b_colK(kp + 0 * kHd + 32, kHd);
    v16bf bk2 = load_b_colK(kp + 16 * kHd, kHd);
    v16bf bk3 = load_b_colK(kp + 16 * kHd + 32, kHd);

    for (int kc = 0; kc < kSeq; kc += 32) {
        v8f s0, s1;
#pragma unroll
        for (int i = 0; i < 8; ++i) { s0[i] = 0.f; s1[i] = 0.f; }
        s0 = wmma_bf16(aq0, bk0, s0);
        s0 = wmma_bf16(aq1, bk1, s0);
        s1 = wmma_bf16(aq0, bk2, s1);
        s1 = wmma_bf16(aq1, bk3, s1);

        // prefetch next chunk's K fragments (overlaps softmax VALU work)
        const int kn = (kc + 32) & (kSeq - 1);
        v16bf nbk0 = load_b_colK(kp + (size_t)kn * kHd, kHd);
        v16bf nbk1 = load_b_colK(kp + (size_t)kn * kHd + 32, kHd);
        v16bf nbk2 = load_b_colK(kp + (size_t)(kn + 16) * kHd, kHd);
        v16bf nbk3 = load_b_colK(kp + (size_t)(kn + 16) * kHd + 32, kHd);

        // online softmax; row r (+8*hi) lives across the 16 lanes of this half-wave
#pragma unroll
        for (int r = 0; r < 8; ++r) {
            float a0 = s0[r] * kScale;
            float a1 = s1[r] * kScale;
            float mx = fmaxf(a0, a1);
            mx = fmaxf(mx, __shfl_xor(mx, 1, 32));
            mx = fmaxf(mx, __shfl_xor(mx, 2, 32));
            mx = fmaxf(mx, __shfl_xor(mx, 4, 32));
            mx = fmaxf(mx, __shfl_xor(mx, 8, 32));
            const float mnew = fmaxf(mrow[r], mx);
            const float alpha = __expf(mrow[r] - mnew);
            const float p0 = __expf(a0 - mnew);
            const float p1 = __expf(a1 - mnew);
            float ps = p0 + p1;
            ps += __shfl_xor(ps, 1, 32);
            ps += __shfl_xor(ps, 2, 32);
            ps += __shfl_xor(ps, 4, 32);
            ps += __shfl_xor(ps, 8, 32);
            lrow[r] = lrow[r] * alpha + ps;
            mrow[r] = mnew;
#pragma unroll
            for (int t = 0; t < 4; ++t) oacc[t][r] *= alpha;
            pbuf[warp][r + 8 * hi][nl]      = f2bf(p0);
            pbuf[warp][r + 8 * hi][nl + 16] = f2bf(p1);
        }

        // V fragments: Vt rows are contiguous along keys -> pure b128 loads
        v16bf bv0 = load_b_colK(vp + (size_t)0  * kSeq + kc, kSeq);
        v16bf bv1 = load_b_colK(vp + (size_t)16 * kSeq + kc, kSeq);
        v16bf bv2 = load_b_colK(vp + (size_t)32 * kSeq + kc, kSeq);
        v16bf bv3 = load_b_colK(vp + (size_t)48 * kSeq + kc, kSeq);

        asm volatile("s_wait_dscnt 0" ::: "memory");  // P stores -> A-frag loads (same wave)
        Frag pf;
        {
            const uint16_t* pp = &pbuf[warp][nl][hi * 8];
            pf.q[0] = *(const uint4*)(pp);
            pf.q[1] = *(const uint4*)(pp + 16);
        }
        oacc[0] = wmma_bf16(pf.v, bv0, oacc[0]);
        oacc[1] = wmma_bf16(pf.v, bv1, oacc[1]);
        oacc[2] = wmma_bf16(pf.v, bv2, oacc[2]);
        oacc[3] = wmma_bf16(pf.v, bv3, oacc[3]);

        bk0 = nbk0; bk1 = nbk1; bk2 = nbk2; bk3 = nbk3;
    }

#pragma unroll
    for (int r = 0; r < 8; ++r) {
        const float inv = 1.0f / lrow[r];
        const int tok = qt * 16 + r + 8 * hi;
        uint16_t* op = O + (size_t)(b * kSeq + tok) * kDim + h * kHd;
#pragma unroll
        for (int t = 0; t < 4; ++t)
            op[t * 16 + nl] = f2bf(oacc[t][r] * inv);
    }
}

// out = A @ Wproj^T + bproj, f32 output
__global__ void __launch_bounds__(256) k_proj(const uint16_t* __restrict__ A,
                                              const uint16_t* __restrict__ W,
                                              const float* __restrict__ bias,
                                              float* __restrict__ out) {
    const int warp = threadIdx.x >> 5, lane = threadIdx.x & 31;
    const int m0 = (blockIdx.x * 8 + warp) * 32;
    const int nb = blockIdx.y * 64;
    GEMM_BODY(A + (size_t)m0 * kDim, W)
    const int hi = lane >> 4, nl = lane & 15;
#pragma unroll
    for (int s = 0; s < 2; ++s)
#pragma unroll
        for (int t = 0; t < 4; ++t) {
            const int col = nb + t * 16 + nl;
            const float bi = bias[col];
#pragma unroll
            for (int r = 0; r < 8; ++r) {
                const int gm = m0 + s * 16 + r + 8 * hi;
                out[(size_t)gm * kDim + col] = acc[s][t][r] + bi;
            }
        }
}

extern "C" void kernel_launch(void* const* d_in, const int* in_sizes, int n_in,
                              void* d_out, int out_size, void* d_ws, size_t ws_size,
                              hipStream_t stream) {
    (void)in_sizes; (void)n_in; (void)out_size; (void)ws_size;
    const float* x     = (const float*)d_in[0];
    const float* Wq    = (const float*)d_in[1];
    const float* Wkv   = (const float*)d_in[2];
    const float* Wproj = (const float*)d_in[3];
    const float* bproj = (const float*)d_in[4];
    float* out = (float*)d_out;

    char* ws = (char*)d_ws;
    size_t off = 0;
    auto alloc = [&](size_t elems) {
        uint16_t* p = (uint16_t*)(ws + off);
        off += elems * sizeof(uint16_t);
        return p;
    };
    uint16_t* Xb   = alloc((size_t)kM * kDim);
    uint16_t* Wqb  = alloc((size_t)kDim * kDim);
    uint16_t* Wkvb = alloc((size_t)512 * kDim);
    uint16_t* Wpb  = alloc((size_t)kDim * kDim);
    uint16_t* Qb   = alloc((size_t)kM * kDim);
    uint16_t* Kb   = alloc((size_t)kBatch * kKV * kSeq * kHd);
    uint16_t* Vtb  = alloc((size_t)kBatch * kKV * kSeq * kHd);
    uint16_t* Ob   = alloc((size_t)kM * kDim);

    auto cvt = [&](const float* src, uint16_t* dst, int n) {
        k_cvt4<<<(n / 4 + 255) / 256, 256, 0, stream>>>((const float4*)src, dst, n / 4);
    };
    cvt(x, Xb, kM * kDim);
    cvt(Wq, Wqb, kDim * kDim);
    cvt(Wkv, Wkvb, 512 * kDim);
    cvt(Wproj, Wpb, kDim * kDim);

    k_gemm_q <<<dim3(kM / 32 / 8, kDim / 64), 256, 0, stream>>>(Xb, Wqb, Qb);
    k_gemm_kv<<<dim3(kM / 32 / 8, 512 / 64),  256, 0, stream>>>(Xb, Wkvb, Kb, Vtb);
    k_attn   <<<(kBatch * kHeads * (kSeq / 16)) / 8, 256, 0, stream>>>(Qb, Kb, Vtb, Ob);
    k_proj   <<<dim3(kM / 32 / 8, kDim / 64), 256, 0, stream>>>(Ob, Wpb, bproj, out);
}
